// CrossAttentionLayer_46523085750536
// MI455X (gfx1250) — matrix operand order repaired and checked by reading, not compiled, not run on hardware
//
#include <hip/hip_runtime.h>
#include <stdint.h>

typedef __bf16 bf16_t;
typedef __attribute__((ext_vector_type(16))) __bf16 v16bf;
typedef __attribute__((ext_vector_type(8)))  float  v8f;
typedef __attribute__((ext_vector_type(4)))  uint32_t u32x4;
typedef __attribute__((ext_vector_type(8)))  uint32_t u32x8;

union FragU {
    v16bf v;
    uint4 q[2];
    __bf16 e[16];
};

union PackU {
    uint32_t u;
    __bf16 e[2];
};

union Pack8U {
    uint4 q;
    __bf16 e[8];
};

static __device__ inline v8f wmma_bf16(const v16bf& a, const v16bf& b, v8f c) {
    // D = A(16x32) * B(32x16) + C, f32 accumulate
    return __builtin_amdgcn_wmma_f32_16x16x32_bf16(false, a, false, b, (short)0, c, false, false);
}

// ---------------------------------------------------------------------------
// Tensor Data Mover: one 2D tile (tile_dim0 x tile_dim1 elements, 2B each)
// global -> LDS, descriptor-driven, tracked by TENSORcnt. Issued per-wave.
// D# layout per CDNA5 ISA ch. 7.3 / 8 (groups 0 and 1; VADDR2/3 omitted for 2D).
// ---------------------------------------------------------------------------
static __device__ inline void tdm_load_tile(uint32_t lds_off, const void* gaddr,
                                            uint32_t rows /*tensor_dim1*/,
                                            uint32_t kdim /*tensor_dim0 = stride*/) {
    uint64_t ga = (uint64_t)(uintptr_t)gaddr;
    u32x4 g0;
    g0[0] = 1u;                                   // count=1, user mode
    g0[1] = lds_off;                              // lds_addr (bytes)
    g0[2] = (uint32_t)ga;                         // global_addr[31:0]
    g0[3] = (uint32_t)((ga >> 32) & 0x1FFFFFFu)   // global_addr[56:32]
            | (2u << 30);                         // type = 2 ("image")
    u32x8 g1;
    g1[0] = 1u << 16;                             // data_size = 1 -> 2 bytes
    g1[1] = (kdim & 0xFFFFu) << 16;               // tensor_dim0[15:0]
    g1[2] = ((kdim >> 16) & 0xFFFFu)              // tensor_dim0[31:16]
            | ((rows & 0xFFFFu) << 16);           // tensor_dim1[15:0]
    g1[3] = ((rows >> 16) & 0xFFFFu)              // tensor_dim1[31:16]
            | (32u << 16);                        // tile_dim0 = 32 (K per step)
    g1[4] = 128u;                                 // tile_dim1 = 128 rows
    g1[5] = kdim;                                 // tensor_dim0_stride[31:0]
    g1[6] = 0u;
    g1[7] = 0u;
    asm volatile("tensor_load_to_lds %0, %1" :: "s"(g0), "s"(g1) : "memory");
}

// ---------------------------------------------------------------------------
// f32 -> bf16 conversion (grid-stride)
// ---------------------------------------------------------------------------
__global__ __launch_bounds__(256) void cvt_f32_bf16(const float* __restrict__ in,
                                                    bf16_t* __restrict__ out, int n) {
    int i = blockIdx.x * blockDim.x + threadIdx.x;
    int stride = gridDim.x * blockDim.x;
    for (; i < n; i += stride) out[i] = (bf16_t)in[i];
}

// ---------------------------------------------------------------------------
// C[M,N] = A[M,K] * W[N,K]^T + bias   (NT GEMM, bf16 in, f32 acc)
// Block: 256 threads (8 waves), tile 128(M) x 128(N), BK=32, M=8192, N=K=1024.
// LDS tiles are filled by the Tensor Data Mover (wave 0 issues one descriptor
// per tile, double-buffered: tile i+1's DMA overlaps WMMA on tile i).
// outMode: 0 -> bf16 [M,N];  1 -> f32 [M,N];
//          2 -> bf16 transposed-per-head V^T layout [b, head, d, s]
//               (B=8, NH=16, HD=64, S=1024 assumed)
// ---------------------------------------------------------------------------
__global__ __launch_bounds__(256) void gemm_nt(const bf16_t* __restrict__ A,
                                               const bf16_t* __restrict__ W,
                                               const float* __restrict__ bias,
                                               void* __restrict__ outp,
                                               int N, int Kd, int outMode) {
    __shared__ __align__(16) bf16_t As[2][128 * 32];
    __shared__ __align__(16) bf16_t Bs[2][128 * 32];

    const int tid  = threadIdx.x;
    const int wid  = tid >> 5;
    const int lane = tid & 31;
    const int lr   = lane & 15;
    const int kh   = lane >> 4;     // K-half of the wave (lanes 0-15 vs 16-31)

    const int m0 = blockIdx.y * 128;
    const int n0 = blockIdx.x * 128;
    const int wm0 = (wid & 3) * 32;   // wave M offset inside tile
    const int wn0 = (wid >> 2) * 64;  // wave N offset inside tile

    const uint32_t baseA = (uint32_t)(uintptr_t)(&As[0][0]);
    const uint32_t baseB = (uint32_t)(uintptr_t)(&Bs[0][0]);

    const v8f zero8 = {0.f, 0.f, 0.f, 0.f, 0.f, 0.f, 0.f, 0.f};
    v8f acc[2][4];
#pragma unroll
    for (int mt = 0; mt < 2; ++mt)
#pragma unroll
        for (int nt = 0; nt < 4; ++nt) acc[mt][nt] = zero8;

    // prologue: TDM fill of buffer 0 (wave 0 only; 2 descriptors in flight)
    if (wid == 0) {
        tdm_load_tile(baseA, A + (size_t)m0 * Kd, 8192u, (uint32_t)Kd);
        tdm_load_tile(baseB, W + (size_t)n0 * Kd, 1024u, (uint32_t)Kd);
    }

    for (int k0 = 0; k0 < Kd; k0 += 32) {
        const int buf = (k0 >> 5) & 1;
        const bool more = (k0 + 32) < Kd;
        if (wid == 0) {
            if (more) {
                const uint32_t bo = (buf ^ 1) * 8192;   // bytes (8KB per buffer)
                tdm_load_tile(baseA + bo, A + (size_t)m0 * Kd + (k0 + 32), 8192u, (uint32_t)Kd);
                tdm_load_tile(baseB + bo, W + (size_t)n0 * Kd + (k0 + 32), 1024u, (uint32_t)Kd);
                __builtin_amdgcn_s_wait_tensorcnt(2);   // current tile's pair done
            } else {
                __builtin_amdgcn_s_wait_tensorcnt(0);
            }
        }
        __syncthreads();       // current buffer visible to all waves

        const bf16_t* ab = &As[buf][0];
        const bf16_t* bb = &Bs[buf][0];
        FragU af[2], bfr[4];
#pragma unroll
        for (int mt = 0; mt < 2; ++mt) {
            int ml = wm0 + mt * 16 + lr;
            const uint4* ar = (const uint4*)(ab + ml * 32);
            af[mt].q[0] = ar[kh];        // k = kh*8 .. kh*8+7
            af[mt].q[1] = ar[2 + kh];    // k = 16+kh*8 ..
        }
#pragma unroll
        for (int nt = 0; nt < 4; ++nt) {
            int nl = wn0 + nt * 16 + lr;
            const uint4* br = (const uint4*)(bb + nl * 32);
            bfr[nt].q[0] = br[kh];          // k = kh*8 .. +7
            bfr[nt].q[1] = br[2 + kh];      // k = 16+kh*8 .. +7
        }
#pragma unroll
        for (int mt = 0; mt < 2; ++mt)
#pragma unroll
            for (int nt = 0; nt < 4; ++nt)
                acc[mt][nt] = wmma_bf16(af[mt].v, bfr[nt].v, acc[mt][nt]);
        __syncthreads();       // done reading buf before TDM refills it
    }

    // ---- epilogue ----
    if (outMode == 2) {
        // V^T scatter: out[((b*16+head)*64+d)*1024 + s], 8 tokens per 16B store
        bf16_t* vt = (bf16_t*)outp;
#pragma unroll
        for (int nt = 0; nt < 4; ++nt) {
            int n = n0 + wn0 + nt * 16 + lr;   // h index
            float bn = bias[n];
            int head = n >> 6, dl = n & 63;
#pragma unroll
            for (int mt = 0; mt < 2; ++mt) {
                int mb = m0 + wm0 + mt * 16 + 8 * kh;  // first of 8 tokens
                int bb2 = mb >> 10, s0 = mb & 1023;
                Pack8U pk;
#pragma unroll
                for (int r = 0; r < 8; ++r) pk.e[r] = (bf16_t)(acc[mt][nt][r] + bn);
                *(uint4*)(vt + ((size_t)((bb2 * 16 + head) * 64 + dl)) * 1024 + s0) = pk.q;
            }
        }
    } else {
#pragma unroll
        for (int nt = 0; nt < 4; ++nt) {
            int n = n0 + wn0 + nt * 16 + lr;
            float bn = bias[n];
#pragma unroll
            for (int mt = 0; mt < 2; ++mt) {
#pragma unroll
                for (int r = 0; r < 8; ++r) {
                    int m = m0 + wm0 + mt * 16 + r + 8 * kh;
                    float val = acc[mt][nt][r] + bn;
                    if (outMode == 1) ((float*)outp)[(size_t)m * N + n] = val;
                    else              ((bf16_t*)outp)[(size_t)m * N + n] = (bf16_t)val;
                }
            }
        }
    }
}

// ---------------------------------------------------------------------------
// Windowed attention, flash-style online softmax. No LDS needed: V comes
// pre-transposed ([b,head,d,s]) so both WMMA operand reads are contiguous.
// One wave per 16-query tile; block = 4 waves. Band: keys in [q0-64, q0+96).
// S^T = K*Q^T (lane owns one query column -> per-lane softmax stats),
// ctx^T = V^T * P^T.
// ---------------------------------------------------------------------------
__global__ __launch_bounds__(128) void attn_win(const bf16_t* __restrict__ Q,
                                                const bf16_t* __restrict__ Kt,
                                                const bf16_t* __restrict__ Vtr,
                                                bf16_t* __restrict__ ctx) {
    constexpr int S = 1024, H = 1024;

    const int tid  = threadIdx.x;
    const int wid  = tid >> 5;
    const int lane = tid & 31;
    const int lr   = lane & 15;
    const int hi   = lane >> 4;
    const int kh   = hi;

    const int tile = blockIdx.x * 4 + wid;
    const int q0   = (tile & 63) << 4;
    const int head = (tile >> 6) & 15;
    const int b    = tile >> 10;

    // Q^T as B-matrix fragments: lane holds query column m=q0+lr, d contiguous.
    const bf16_t* Qp = Q + ((size_t)(b * S + q0 + lr)) * H + head * 64;
    FragU qf[2];
#pragma unroll
    for (int f = 0; f < 2; ++f) {
        const bf16_t* base = Qp + 32 * f + kh * 8;
        qf[f].q[0] = *(const uint4*)(base);
        qf[f].q[1] = *(const uint4*)(base + 16);
    }

    // V^T rows for this (b, head): [d][s], s contiguous
    const bf16_t* Vh = Vtr + ((size_t)(b * 16 + head) * 64) * 1024;

    const v8f zero8 = {0.f, 0.f, 0.f, 0.f, 0.f, 0.f, 0.f, 0.f};
    v8f acc[4];
#pragma unroll
    for (int c = 0; c < 4; ++c) acc[c] = zero8;

    float mrun = -1e30f, lrun = 0.f;
    const int iq = q0 + lr;

    for (int p = 0; p < 5; ++p) {
        const int jb = q0 - 64 + p * 32;

        // ---- S^T tiles (2 x 16 keys) ----
        float sc[2][8];
#pragma unroll
        for (int tt = 0; tt < 2; ++tt) {
            int key = jb + tt * 16 + lr;
            int kc  = key < 0 ? 0 : (key > S - 1 ? S - 1 : key);
            const bf16_t* Kp = Kt + ((size_t)(b * S + kc)) * H + head * 64;
            FragU ka0, ka1;
            ka0.q[0] = *(const uint4*)(Kp + kh * 8);
            ka0.q[1] = *(const uint4*)(Kp + 16 + kh * 8);
            ka1.q[0] = *(const uint4*)(Kp + 32 + kh * 8);
            ka1.q[1] = *(const uint4*)(Kp + 48 + kh * 8);
            v8f st = zero8;
            st = wmma_bf16(ka0.v, qf[0].v, st);
            st = wmma_bf16(ka1.v, qf[1].v, st);
#pragma unroll
            for (int r = 0; r < 8; ++r) {
                int j = jb + tt * 16 + r + 8 * hi;
                bool ok = (j >= 0) && (j < S) && (j - iq <= 64) && (iq - j <= 64);
                sc[tt][r] = ok ? st[r] * 0.125f : -1e30f;
            }
        }

        // ---- online softmax update (per-lane column stats + one xor-16) ----
        float lm = -1e30f;
#pragma unroll
        for (int tt = 0; tt < 2; ++tt)
#pragma unroll
            for (int r = 0; r < 8; ++r) lm = fmaxf(lm, sc[tt][r]);
        lm = fmaxf(lm, __shfl_xor(lm, 16, 32));
        float nm   = fmaxf(mrun, lm);
        float corr = __expf(mrun - nm);
        float pt[2][8];
        float ps = 0.f;
#pragma unroll
        for (int tt = 0; tt < 2; ++tt)
#pragma unroll
            for (int r = 0; r < 8; ++r) {
                pt[tt][r] = __expf(sc[tt][r] - nm);
                ps += pt[tt][r];
            }
        ps += __shfl_xor(ps, 16, 32);
        lrun = lrun * corr + ps;
        mrun = nm;
#pragma unroll
        for (int c = 0; c < 4; ++c)
#pragma unroll
            for (int r = 0; r < 8; ++r) acc[c][r] *= corr;

        // ---- build P^T as B-matrix (32 keys K-dim x 16 queries) ----
        FragU b2;
        float recv[8];
#pragma unroll
        for (int r = 0; r < 8; ++r) {
            float send = hi ? pt[0][r] : pt[1][r];
            recv[r] = __shfl_xor(send, 16, 32);
        }
        if (!hi) {
#pragma unroll
            for (int h = 0; h < 8; ++h) {
                b2.e[h]     = (bf16_t)pt[0][h];   // tile0 jloc 0..7
                b2.e[8 + h] = (bf16_t)recv[h];    // tile0 jloc 8..15 (from hi partner)
            }
        } else {
#pragma unroll
            for (int h = 0; h < 8; ++h) {
                b2.e[h]     = (bf16_t)recv[h];    // tile1 jloc 0..7 (from lo partner)
                b2.e[8 + h] = (bf16_t)pt[1][h];   // tile1 jloc 8..15
            }
        }

        // ---- ctx^T += V^T * P^T over 4 dim-chunks (contiguous V^T loads) ----
        int st0 = jb + kh * 16;       // A-layout: e[0..7]  <-> j = kh*16+h
        int st1 = jb + kh * 16 + 8;   //           e[8..15] <-> j = kh*16+8+h
        if (st0 < 0 || st0 > S - 8) st0 = 0;   // fully-invalid runs (P=0) -> safe row
        if (st1 < 0 || st1 > S - 8) st1 = 0;
#pragma unroll
        for (int c = 0; c < 4; ++c) {
            const bf16_t* vrow = Vh + (size_t)(16 * c + lr) * 1024;
            FragU a2;
            a2.q[0] = *(const uint4*)(vrow + st0);
            a2.q[1] = *(const uint4*)(vrow + st1);
            acc[c] = wmma_bf16(a2.v, b2.v, acc[c]);
        }
    }

    // ---- finalize: divide by l, write ctx[b, q0+m, head*64+d] as bf16 ----
    float inv = 1.0f / lrun;
    bf16_t* cp = ctx + ((size_t)(b * S + q0 + lr)) * H + head * 64;
#pragma unroll
    for (int c = 0; c < 4; ++c) {
#pragma unroll
        for (int r = 0; r < 8; r += 2) {
            PackU pk;
            pk.e[0] = (bf16_t)(acc[c][r] * inv);
            pk.e[1] = (bf16_t)(acc[c][r + 1] * inv);
            *(uint32_t*)(cp + 16 * c + 8 * hi + r) = pk.u;
        }
    }
}

// ---------------------------------------------------------------------------
// Fused sigmoid gate + blend + LayerNorm. One block (256 thr) per row of 1024.
// out = LN(0.5*hs + 0.5*sigmoid(gl)*ao) * g + b
// ---------------------------------------------------------------------------
__global__ __launch_bounds__(256) void fuse_ln(const float* __restrict__ hs,
                                               const float* __restrict__ gl,
                                               const float* __restrict__ ao,
                                               const float* __restrict__ lng,
                                               const float* __restrict__ lnb,
                                               float* __restrict__ out) {
    const size_t base = (size_t)blockIdx.x * 1024;
    const int tid = threadIdx.x;
    const int wid = tid >> 5;
    const int lane = tid & 31;

    float x[4];
    float s1 = 0.f, s2 = 0.f;
#pragma unroll
    for (int j = 0; j < 4; ++j) {
        int col = tid + j * 256;
        float h = hs[base + col];
        float g = 1.0f / (1.0f + __expf(-gl[base + col]));
        float v = 0.5f * h + 0.5f * (g * ao[base + col]);
        x[j] = v;
        s1 += v;
        s2 += v * v;
    }
#pragma unroll
    for (int off = 16; off > 0; off >>= 1) {
        s1 += __shfl_xor(s1, off, 32);
        s2 += __shfl_xor(s2, off, 32);
    }
    __shared__ float r1[8], r2[8];
    if (lane == 0) { r1[wid] = s1; r2[wid] = s2; }
    __syncthreads();
    s1 = 0.f; s2 = 0.f;
#pragma unroll
    for (int i = 0; i < 8; ++i) { s1 += r1[i]; s2 += r2[i]; }

    float mean = s1 * (1.0f / 1024.0f);
    float var  = s2 * (1.0f / 1024.0f) - mean * mean;
    float rs   = rsqrtf(var + 1e-5f);
#pragma unroll
    for (int j = 0; j < 4; ++j) {
        int col = tid + j * 256;
        out[base + col] = (x[j] - mean) * rs * lng[col] + lnb[col];
    }
}

// ---------------------------------------------------------------------------
extern "C" void kernel_launch(void* const* d_in, const int* in_sizes, int n_in,
                              void* d_out, int out_size, void* d_ws, size_t ws_size,
                              hipStream_t stream) {
    const float* hs  = (const float*)d_in[0];
    const float* cs  = (const float*)d_in[1];
    const float* Wq  = (const float*)d_in[2];
    const float* bq  = (const float*)d_in[3];
    const float* Wk  = (const float*)d_in[4];
    const float* bk  = (const float*)d_in[5];
    const float* Wv  = (const float*)d_in[6];
    const float* bv  = (const float*)d_in[7];
    const float* Wo  = (const float*)d_in[8];
    const float* bo  = (const float*)d_in[9];
    const float* Wg  = (const float*)d_in[10];
    const float* bg  = (const float*)d_in[11];
    const float* lng = (const float*)d_in[12];
    const float* lnb = (const float*)d_in[13];

    const size_t MB = 1024ull * 1024ull;
    char* ws = (char*)d_ws;
    bf16_t* hsb   = (bf16_t*)(ws + 0 * MB);     // 16 MB
    bf16_t* csb   = (bf16_t*)(ws + 16 * MB);    // 16 MB
    bf16_t* wqb   = (bf16_t*)(ws + 32 * MB);    // 2 MB
    bf16_t* wkb   = (bf16_t*)(ws + 34 * MB);    // 2 MB
    bf16_t* wvb   = (bf16_t*)(ws + 36 * MB);    // 2 MB
    bf16_t* wgb   = (bf16_t*)(ws + 38 * MB);    // 2 MB
    bf16_t* wob   = (bf16_t*)(ws + 40 * MB);    // 2 MB
    bf16_t* Qb    = (bf16_t*)(ws + 42 * MB);    // 16 MB
    bf16_t* Kb    = (bf16_t*)(ws + 58 * MB);    // 16 MB
    bf16_t* Vtr   = (bf16_t*)(ws + 74 * MB);    // 16 MB (transposed-per-head V)
    bf16_t* ctxb  = (bf16_t*)(ws + 90 * MB);    // 16 MB
    float*  gateL = (float*)(ws + 106 * MB);    // 32 MB
    float*  aout  = (float*)(ws + 138 * MB);    // 32 MB  (total 170 MB)

    const int NTOK = 8192;   // B*S
    const int Hn   = 1024;

    cvt_f32_bf16<<<2048, 256, 0, stream>>>(hs, hsb, NTOK * Hn);
    cvt_f32_bf16<<<2048, 256, 0, stream>>>(cs, csb, NTOK * Hn);
    cvt_f32_bf16<<<512,  256, 0, stream>>>(Wq, wqb, Hn * Hn);
    cvt_f32_bf16<<<512,  256, 0, stream>>>(Wk, wkb, Hn * Hn);
    cvt_f32_bf16<<<512,  256, 0, stream>>>(Wv, wvb, Hn * Hn);
    cvt_f32_bf16<<<512,  256, 0, stream>>>(Wg, wgb, Hn * Hn);
    cvt_f32_bf16<<<512,  256, 0, stream>>>(Wo, wob, Hn * Hn);

    dim3 gg(Hn / 128, NTOK / 128);   // (8, 64)
    gemm_nt<<<gg, 256, 0, stream>>>(hsb, wqb, bq, (void*)Qb,    Hn, Hn, 0);
    gemm_nt<<<gg, 256, 0, stream>>>(csb, wkb, bk, (void*)Kb,    Hn, Hn, 0);
    gemm_nt<<<gg, 256, 0, stream>>>(csb, wvb, bv, (void*)Vtr,   Hn, Hn, 2);
    gemm_nt<<<gg, 256, 0, stream>>>(hsb, wgb, bg, (void*)gateL, Hn, Hn, 1);

    attn_win<<<2048, 128, 0, stream>>>(Qb, Kb, Vtr, ctxb);

    gemm_nt<<<gg, 256, 0, stream>>>(ctxb, wob, bo, (void*)aout, Hn, Hn, 1);

    fuse_ln<<<NTOK, 256, 0, stream>>>(hs, gateL, aout, lng, lnb, (float*)d_out);
}
